// GNN_node_Virtualnode_28432683499899
// MI455X (gfx1250) — matrix operand order repaired and checked by reading, not compiled, not run on hardware
//
#include <hip/hip_runtime.h>
#include <hip/hip_bf16.h>

#define N_NODES 100000
#define N_EDGES 600000
#define G_GRAPHS 2000

typedef __attribute__((ext_vector_type(8)))  float  v8f;
typedef __bf16 v16bf __attribute__((ext_vector_type(16)));

union BF16x16 {
    v16bf v;
    uint4 q[2];
};

// Native bf16 convert: fptrunc f32 -> bf16 (RNE). Backend should select the
// hardware convert on gfx1250.
__device__ __forceinline__ __bf16 f2bf(float f) { return (__bf16)f; }

// ---- fragment loaders ----------------------------------------------------
// A fragment (16x32 bf16, MxK): lane m=l&15 covers K = kb..kb+7 and
// kb+16..kb+23 where kb = (l>=16)*8.  p points at row + (kk + kb).
__device__ __forceinline__ v16bf load_a_frag(const float* __restrict__ p) {
    const float4* ap = reinterpret_cast<const float4*>(p);
    float4 f0 = ap[0], f1 = ap[1], f2 = ap[4], f3 = ap[5];
    v16bf a;
    a[0]  = f2bf(f0.x);  a[1]  = f2bf(f0.y);  a[2]  = f2bf(f0.z);  a[3]  = f2bf(f0.w);
    a[4]  = f2bf(f1.x);  a[5]  = f2bf(f1.y);  a[6]  = f2bf(f1.z);  a[7]  = f2bf(f1.w);
    a[8]  = f2bf(f2.x);  a[9]  = f2bf(f2.y);  a[10] = f2bf(f2.z);  a[11] = f2bf(f2.w);
    a[12] = f2bf(f3.x);  a[13] = f2bf(f3.y);  a[14] = f2bf(f3.z);  a[15] = f2bf(f3.w);
    return a;
}

// A fragment with fused batchnorm + relu (scale/shift indexed by K channel).
__device__ __forceinline__ v16bf load_a_frag_bn(const float* __restrict__ p,
                                                const float* __restrict__ sc,
                                                const float* __restrict__ sh) {
    const float4* ap = reinterpret_cast<const float4*>(p);
    float4 f0 = ap[0], f1 = ap[1], f2 = ap[4], f3 = ap[5];
    float fv[16] = {f0.x,f0.y,f0.z,f0.w, f1.x,f1.y,f1.z,f1.w,
                    f2.x,f2.y,f2.z,f2.w, f3.x,f3.y,f3.z,f3.w};
    v16bf a;
#pragma unroll
    for (int i = 0; i < 8; i++)
        a[i] = f2bf(fmaxf(fv[i] * sc[i] + sh[i], 0.f));
#pragma unroll
    for (int i = 0; i < 8; i++)
        a[8 + i] = f2bf(fmaxf(fv[8 + i] * sc[16 + i] + sh[16 + i], 0.f));
    return a;
}

// B fragment (32x16 bf16, KxN): lane n=l&15 covers K = kb..kb+7 and
// kb+16..kb+23 where kb = (l>=16)*8.  Weights are stored transposed [Ncol][K] so this is
// two 16-byte loads.
__device__ __forceinline__ v16bf load_b_frag(const __bf16* __restrict__ p) {
    const uint4* q = reinterpret_cast<const uint4*>(p);
    BF16x16 b;
    b.q[0] = q[0];
    b.q[1] = q[2];
    return b.v;
}

// ---- GEMM1: z1[N,256] = agg[N,128] @ W1 + b1, fused BN stat accumulation --
__global__ __launch_bounds__(128) void k_gemm1(const float* __restrict__ A,
                                               const __bf16* __restrict__ Bt,  // [256][128]
                                               const float* __restrict__ bias, // [256]
                                               float* __restrict__ C,          // [N,256]
                                               float* __restrict__ stat) {     // sum@0 sumsq@256
    const int lane = threadIdx.x & 31;
    const int wave = threadIdx.x >> 5;      // 0..3
    const int row0 = blockIdx.x * 16;
    const int colBase = wave * 64;
    const int m  = lane & 15;
    const int hi = lane >> 4;               // 0 or 1
    v8f c0 = {}, c1 = {}, c2 = {}, c3 = {};
    const float* aRow = A + (size_t)(row0 + m) * 128 + hi * 8;
    const __bf16* bp0 = Bt + (size_t)(colBase +  0 + m) * 128 + hi * 8;
    const __bf16* bp1 = Bt + (size_t)(colBase + 16 + m) * 128 + hi * 8;
    const __bf16* bp2 = Bt + (size_t)(colBase + 32 + m) * 128 + hi * 8;
    const __bf16* bp3 = Bt + (size_t)(colBase + 48 + m) * 128 + hi * 8;
#pragma unroll
    for (int kk = 0; kk < 128; kk += 32) {
        v16bf a = load_a_frag(aRow + kk);
        c0 = __builtin_amdgcn_wmma_f32_16x16x32_bf16(false, a, false, load_b_frag(bp0 + kk), (short)0, c0, false, false);
        c1 = __builtin_amdgcn_wmma_f32_16x16x32_bf16(false, a, false, load_b_frag(bp1 + kk), (short)0, c1, false, false);
        c2 = __builtin_amdgcn_wmma_f32_16x16x32_bf16(false, a, false, load_b_frag(bp2 + kk), (short)0, c2, false, false);
        c3 = __builtin_amdgcn_wmma_f32_16x16x32_bf16(false, a, false, load_b_frag(bp3 + kk), (short)0, c3, false, false);
    }
    v8f acc[4] = {c0, c1, c2, c3};
#pragma unroll
    for (int t = 0; t < 4; t++) {
        int col = colBase + t * 16 + m;
        float bv = bias[col];
        float s = 0.f, s2 = 0.f;
        float* cp = C + (size_t)(row0 + hi * 8) * 256 + col;
#pragma unroll
        for (int i = 0; i < 8; i++) {
            float v = acc[t][i] + bv;
            cp[(size_t)i * 256] = v;
            s += v; s2 += v * v;
        }
        s  += __shfl_xor(s, 16, 32);
        s2 += __shfl_xor(s2, 16, 32);
        if (lane < 16) {
            atomicAdd(&stat[col], s);
            atomicAdd(&stat[256 + col], s2);
        }
    }
}

// ---- GEMM2: z2[N,128] = relu(bn(z1)) @ W2 + b2, fused BN stats ------------
__global__ __launch_bounds__(128) void k_gemm2(const float* __restrict__ A,    // z1 [N,256]
                                               const float* __restrict__ bnA,  // scale@512 shift@768
                                               const __bf16* __restrict__ Bt,  // [128][256]
                                               const float* __restrict__ bias, // [128]
                                               float* __restrict__ C,          // [N,128]
                                               float* __restrict__ stat) {
    const int lane = threadIdx.x & 31;
    const int wave = threadIdx.x >> 5;
    const int row0 = blockIdx.x * 16;
    const int colBase = wave * 32;
    const int m  = lane & 15;
    const int hi = lane >> 4;
    const float* scale = bnA + 512;
    const float* shift = bnA + 768;
    v8f c0 = {}, c1 = {};
    const float* aRow = A + (size_t)(row0 + m) * 256 + hi * 8;
    const __bf16* bp0 = Bt + (size_t)(colBase +  0 + m) * 256 + hi * 8;
    const __bf16* bp1 = Bt + (size_t)(colBase + 16 + m) * 256 + hi * 8;
#pragma unroll
    for (int kk = 0; kk < 256; kk += 32) {
        int kb = kk + hi * 8;
        v16bf a = load_a_frag_bn(aRow + kk, scale + kb, shift + kb);
        c0 = __builtin_amdgcn_wmma_f32_16x16x32_bf16(false, a, false, load_b_frag(bp0 + kk), (short)0, c0, false, false);
        c1 = __builtin_amdgcn_wmma_f32_16x16x32_bf16(false, a, false, load_b_frag(bp1 + kk), (short)0, c1, false, false);
    }
    v8f acc[2] = {c0, c1};
#pragma unroll
    for (int t = 0; t < 2; t++) {
        int col = colBase + t * 16 + m;
        float bv = bias[col];
        float s = 0.f, s2 = 0.f;
        float* cp = C + (size_t)(row0 + hi * 8) * 128 + col;
#pragma unroll
        for (int i = 0; i < 8; i++) {
            float v = acc[t][i] + bv;
            cp[(size_t)i * 128] = v;
            s += v; s2 += v * v;
        }
        s  += __shfl_xor(s, 16, 32);
        s2 += __shfl_xor(s2, 16, 32);
        if (lane < 16) {
            atomicAdd(&stat[col], s);
            atomicAdd(&stat[256 + col], s2);
        }
    }
}

// ---- elementwise / graph kernels ------------------------------------------
__global__ void k_atom(const int* __restrict__ x, const float* __restrict__ emb,
                       float* __restrict__ h) {
    int t = blockIdx.x * 256 + threadIdx.x;   // exactly N*128 threads
    int n = t >> 7, d = t & 127;
    const int* xr = x + n * 9;
    float s = 0.f;
#pragma unroll
    for (int k = 0; k < 9; k++) s += emb[(size_t)(k * 64 + xr[k]) * 128 + d];
    h[t] = s;
}

__global__ void k_vninit(const float* __restrict__ vn_emb, float* __restrict__ vn) {
    int t = blockIdx.x * 256 + threadIdx.x;   // G*128
    vn[t] = vn_emb[t & 127];
}

__global__ void k_addvn(float* __restrict__ h, const float* __restrict__ vn,
                        const int* __restrict__ batch) {
    int t = blockIdx.x * 256 + threadIdx.x;   // N*128
    h[t] += vn[(size_t)batch[t >> 7] * 128 + (t & 127)];
}

__global__ void k_initagg(const float* __restrict__ h, const float* __restrict__ eps,
                          int l, float* __restrict__ agg) {
    int t = blockIdx.x * 256 + threadIdx.x;   // N*128
    agg[t] = (1.0f + eps[l]) * h[t];
}

__global__ __launch_bounds__(256) void k_edges(const float* __restrict__ h,
                                               const int* __restrict__ ei,
                                               const int* __restrict__ ea,
                                               const float* __restrict__ bemb, // [3][8][128] layer slice
                                               float* __restrict__ agg) {
    int t = blockIdx.x * 256 + threadIdx.x;   // E*32 threads
    int e = t >> 5;
    int c = (t & 31) << 2;
    int src = ei[e], dst = ei[N_EDGES + e];
    int a0 = ea[e * 3], a1 = ea[e * 3 + 1], a2 = ea[e * 3 + 2];
    float4 hv = *reinterpret_cast<const float4*>(h + (size_t)src * 128 + c);
    float4 b0 = *reinterpret_cast<const float4*>(bemb + (size_t)(0 * 8 + a0) * 128 + c);
    float4 b1 = *reinterpret_cast<const float4*>(bemb + (size_t)(1 * 8 + a1) * 128 + c);
    float4 b2 = *reinterpret_cast<const float4*>(bemb + (size_t)(2 * 8 + a2) * 128 + c);
    float m0 = fmaxf(hv.x + b0.x + b1.x + b2.x, 0.f);
    float m1 = fmaxf(hv.y + b0.y + b1.y + b2.y, 0.f);
    float m2 = fmaxf(hv.z + b0.z + b1.z + b2.z, 0.f);
    float m3 = fmaxf(hv.w + b0.w + b1.w + b2.w, 0.f);
    float* o = agg + (size_t)dst * 128 + c;
    atomicAdd(o + 0, m0); atomicAdd(o + 1, m1);
    atomicAdd(o + 2, m2); atomicAdd(o + 3, m3);
}

__global__ void k_zero(float* __restrict__ p, int n) {
    int i = blockIdx.x * 256 + threadIdx.x;
    if (i < n) p[i] = 0.f;
}

__global__ void k_bnfin(float* __restrict__ bn, const float* __restrict__ g,
                        const float* __restrict__ b, int ch, float invn) {
    int c = threadIdx.x;
    if (c >= ch) return;
    float mean = bn[c] * invn;
    float var  = bn[256 + c] * invn - mean * mean;
    float inv  = rsqrtf(var + 1e-5f);
    float sc   = g[c] * inv;
    bn[512 + c] = sc;
    bn[768 + c] = b[c] - mean * sc;
}

__global__ void k_copy(const float* __restrict__ src, float* __restrict__ dst) {
    int t = blockIdx.x * 256 + threadIdx.x;
    dst[t] = src[t];
}

__global__ void k_pool(const float* __restrict__ h, const int* __restrict__ batch,
                       float* __restrict__ vnp) {
    int t = blockIdx.x * 256 + threadIdx.x;   // N*128
    atomicAdd(&vnp[(size_t)batch[t >> 7] * 128 + (t & 127)], h[t]);
}

__global__ void k_vng1(const float* __restrict__ vnp, const float* __restrict__ W,
                       const float* __restrict__ b, float* __restrict__ t1,
                       float* __restrict__ bn) {
    int t = blockIdx.x * 256 + threadIdx.x;   // G*256
    int g = t >> 8, j = t & 255;
    const float* vr = vnp + (size_t)g * 128;
    float acc = b[j];
    for (int k = 0; k < 128; k++) acc += vr[k] * W[(size_t)k * 256 + j];
    t1[t] = acc;
    atomicAdd(&bn[j], acc);
    atomicAdd(&bn[256 + j], acc * acc);
}

__global__ void k_vng2(const float* __restrict__ t1, const float* __restrict__ bnV1,
                       const float* __restrict__ W, const float* __restrict__ b,
                       float* __restrict__ t2, float* __restrict__ bn) {
    int t = blockIdx.x * 256 + threadIdx.x;   // G*128
    int g = t >> 7, j = t & 127;
    const float* tr = t1 + (size_t)g * 256;
    const float* sc = bnV1 + 512;
    const float* sh = bnV1 + 768;
    float acc = b[j];
    for (int k = 0; k < 256; k++) {
        float v = fmaxf(tr[k] * sc[k] + sh[k], 0.f);
        acc += v * W[(size_t)k * 128 + j];
    }
    t2[t] = acc;
    atomicAdd(&bn[j], acc);
    atomicAdd(&bn[256 + j], acc * acc);
}

__global__ void k_vnapply(const float* __restrict__ t2, const float* __restrict__ bnV2,
                          float* __restrict__ vn) {
    int t = blockIdx.x * 256 + threadIdx.x;   // G*128
    int d = t & 127;
    vn[t] = fmaxf(t2[t] * bnV2[512 + d] + bnV2[768 + d], 0.f);
}

__global__ void k_hupd(const float* __restrict__ z2, const float* __restrict__ bn,
                       float* __restrict__ out, int dorelu) {
    int t = blockIdx.x * 256 + threadIdx.x;   // N*128
    int d = t & 127;
    float v = z2[t] * bn[512 + d] + bn[768 + d];
    if (dorelu) v = fmaxf(v, 0.f);
    out[t] = v;
}

// weights -> transposed bf16
__global__ void k_convw1(const float* __restrict__ W, __bf16* __restrict__ Wt) {
    int t = blockIdx.x * 256 + threadIdx.x;   // 5*128*256
    int l = t >> 15, r = t & 32767;
    int k = r >> 8, j = r & 255;
    Wt[(size_t)l * 32768 + (size_t)j * 128 + k] = f2bf(W[t]);
}

__global__ void k_convw2(const float* __restrict__ W, __bf16* __restrict__ Wt) {
    int t = blockIdx.x * 256 + threadIdx.x;   // 5*256*128
    int l = t >> 15, r = t & 32767;
    int k = r >> 7, j = r & 127;
    Wt[(size_t)l * 32768 + (size_t)j * 256 + k] = f2bf(W[t]);
}

extern "C" void kernel_launch(void* const* d_in, const int* in_sizes, int n_in,
                              void* d_out, int out_size, void* d_ws, size_t ws_size,
                              hipStream_t stream) {
    const int*   x        = (const int*)d_in[0];
    const int*   ei       = (const int*)d_in[1];
    const int*   ea       = (const int*)d_in[2];
    const int*   batch    = (const int*)d_in[3];
    const float* atom_emb = (const float*)d_in[4];
    const float* vn_emb   = (const float*)d_in[5];
    const float* bond_emb = (const float*)d_in[6];
    const float* eps      = (const float*)d_in[7];
    const float* cW1      = (const float*)d_in[8];
    const float* cb1      = (const float*)d_in[9];
    const float* cbng     = (const float*)d_in[10];
    const float* cbnb     = (const float*)d_in[11];
    const float* cW2      = (const float*)d_in[12];
    const float* cb2      = (const float*)d_in[13];
    const float* bng      = (const float*)d_in[14];
    const float* bnb      = (const float*)d_in[15];
    const float* vW1      = (const float*)d_in[16];
    const float* vb1      = (const float*)d_in[17];
    const float* vbn1g    = (const float*)d_in[18];
    const float* vbn1b    = (const float*)d_in[19];
    const float* vW2      = (const float*)d_in[20];
    const float* vb2      = (const float*)d_in[21];
    const float* vbn2g    = (const float*)d_in[22];
    const float* vbn2b    = (const float*)d_in[23];
    float* out = (float*)d_out;

    float* ws   = (float*)d_ws;
    float* h    = ws;                                  // N*128
    float* agg  = h   + (size_t)N_NODES * 128;         // N*128 (reused as z2)
    float* z1   = agg + (size_t)N_NODES * 128;         // N*256
    float* vn   = z1  + (size_t)N_NODES * 256;         // G*128
    float* vnp  = vn  + (size_t)G_GRAPHS * 128;        // G*128
    float* t1   = vnp + (size_t)G_GRAPHS * 128;        // G*256
    float* t2   = t1  + (size_t)G_GRAPHS * 256;        // G*128
    float* bnA  = t2  + (size_t)G_GRAPHS * 128;        // 1024
    float* bnB  = bnA + 1024;                          // 1024
    float* bnV1 = bnB + 1024;                          // 1024
    float* bnV2 = bnV1 + 1024;                         // 1024
    __bf16* w1t = (__bf16*)(bnV2 + 1024);              // 5*256*128 bf16
    __bf16* w2t = w1t + 5 * 32768;                     // 5*128*256 bf16

    const int NB  = (N_NODES * 128) / 256;   // 50000
    const int EB  = (N_EDGES * 32) / 256;    // 75000
    const int GB  = (G_GRAPHS * 128) / 256;  // 1000
    const int MTB = N_NODES / 16;            // 6250

    k_convw1<<<640, 256, 0, stream>>>(cW1, w1t);
    k_convw2<<<640, 256, 0, stream>>>(cW2, w2t);
    k_atom<<<NB, 256, 0, stream>>>(x, atom_emb, h);
    k_vninit<<<GB, 256, 0, stream>>>(vn_emb, vn);

    for (int l = 0; l < 5; l++) {
        k_addvn<<<NB, 256, 0, stream>>>(h, vn, batch);
        k_initagg<<<NB, 256, 0, stream>>>(h, eps, l, agg);
        k_edges<<<EB, 256, 0, stream>>>(h, ei, ea,
                                        bond_emb + (size_t)l * 3 * 8 * 128, agg);
        k_zero<<<2, 256, 0, stream>>>(bnA, 512);
        k_gemm1<<<MTB, 128, 0, stream>>>(agg, w1t + (size_t)l * 32768,
                                         cb1 + l * 256, z1, bnA);
        k_bnfin<<<1, 256, 0, stream>>>(bnA, cbng + l * 256, cbnb + l * 256,
                                       256, 1.0f / (float)N_NODES);
        k_zero<<<1, 256, 0, stream>>>(bnB, 512);
        k_gemm2<<<MTB, 128, 0, stream>>>(z1, bnA, w2t + (size_t)l * 32768,
                                         cb2 + l * 128, agg, bnB);
        k_bnfin<<<1, 128, 0, stream>>>(bnB, bng + l * 128, bnb + l * 128,
                                       128, 1.0f / (float)N_NODES);
        if (l < 4) {
            // virtual node update uses pre-conv h
            k_copy<<<GB, 256, 0, stream>>>(vn, vnp);
            k_pool<<<NB, 256, 0, stream>>>(h, batch, vnp);
            k_zero<<<2, 256, 0, stream>>>(bnV1, 512);
            k_vng1<<<2000, 256, 0, stream>>>(vnp, vW1 + (size_t)l * 128 * 256,
                                             vb1 + l * 256, t1, bnV1);
            k_bnfin<<<1, 256, 0, stream>>>(bnV1, vbn1g + l * 256, vbn1b + l * 256,
                                           256, 1.0f / (float)G_GRAPHS);
            k_zero<<<1, 256, 0, stream>>>(bnV2, 512);
            k_vng2<<<GB, 256, 0, stream>>>(t1, bnV1, vW2 + (size_t)l * 256 * 128,
                                           vb2 + l * 128, t2, bnV2);
            k_bnfin<<<1, 128, 0, stream>>>(bnV2, vbn2g + l * 128, vbn2b + l * 128,
                                           128, 1.0f / (float)G_GRAPHS);
            k_vnapply<<<GB, 256, 0, stream>>>(t2, bnV2, vn);
            k_hupd<<<NB, 256, 0, stream>>>(agg, bnB, h, 1);
        } else {
            k_hupd<<<NB, 256, 0, stream>>>(agg, bnB, out, 0);
        }
    }
}
